// GINE_Net_65000035058412
// MI455X (gfx1250) — compile-verified, matrix operand order checked by reading, not run on hardware
//
#include <hip/hip_runtime.h>
#include <hip/hip_bf16.h>

#define N_NODES 50000
#define N_EDGES 500000
#define N_GRAPHS 1000
#define HID 128
#define OUTD 10
#define NLAYERS 4

typedef __attribute__((ext_vector_type(16))) __bf16 v16bf;
typedef __attribute__((ext_vector_type(8)))  __bf16 v8bf;
typedef __attribute__((ext_vector_type(8)))  float  v8f;

// ---------------------------------------------------------------------------
// AtomEncoder: h[n,c] = sum_f atom_emb[f, x[n,f], c]
// one thread = one (node, 4-channel chunk)
// ---------------------------------------------------------------------------
__global__ __launch_bounds__(256) void k_encode_nodes(
    const int* __restrict__ x, const float* __restrict__ atom_emb,
    float* __restrict__ h)
{
    int gid = blockIdx.x * 256 + threadIdx.x;
    if (gid >= N_NODES * 32) return;
    int n = gid >> 5;
    int c = (gid & 31) * 4;
    float4 acc = make_float4(0.f, 0.f, 0.f, 0.f);
    #pragma unroll
    for (int f = 0; f < 9; ++f) {
        int idx = x[n * 9 + f];
        const float4 v = *(const float4*)(atom_emb + ((size_t)(f * 120 + idx)) * HID + c);
        acc.x += v.x; acc.y += v.y; acc.z += v.z; acc.w += v.w;
    }
    *(float4*)(h + (size_t)n * HID + c) = acc;
}

// ---------------------------------------------------------------------------
// agg = h  (self term of z = h + sum of messages)
// ---------------------------------------------------------------------------
__global__ __launch_bounds__(256) void k_copy_h(
    const float* __restrict__ h, float* __restrict__ agg)
{
    int gid = blockIdx.x * 256 + threadIdx.x;
    if (gid >= N_NODES * 32) return;
    ((float4*)agg)[gid] = ((const float4*)h)[gid];
}

// ---------------------------------------------------------------------------
// Edge messages: agg[dst] += relu(h[src] + e), e recomputed from 9KB bond table
// one thread = one (edge, 4-channel chunk); 4x global_atomic_add_f32
// ---------------------------------------------------------------------------
__global__ __launch_bounds__(256) void k_edges(
    const int* __restrict__ edge_index, const int* __restrict__ edge_attr,
    const float* __restrict__ bond_emb, const float* __restrict__ h,
    float* __restrict__ agg)
{
    int gid = blockIdx.x * 256 + threadIdx.x;
    if (gid >= N_EDGES * 32) return;
    int e = gid >> 5;
    int c = (gid & 31) * 4;
    int src = edge_index[e];
    int dst = edge_index[N_EDGES + e];
    int a0 = edge_attr[e * 3 + 0];
    int a1 = edge_attr[e * 3 + 1];
    int a2 = edge_attr[e * 3 + 2];
    const float4 e0 = *(const float4*)(bond_emb + ((size_t)(0 * 6 + a0)) * HID + c);
    const float4 e1 = *(const float4*)(bond_emb + ((size_t)(1 * 6 + a1)) * HID + c);
    const float4 e2 = *(const float4*)(bond_emb + ((size_t)(2 * 6 + a2)) * HID + c);
    const float4 hv = *(const float4*)(h + (size_t)src * HID + c);
    float m0 = hv.x + e0.x + e1.x + e2.x; m0 = m0 > 0.f ? m0 : 0.f;
    float m1 = hv.y + e0.y + e1.y + e2.y; m1 = m1 > 0.f ? m1 : 0.f;
    float m2 = hv.z + e0.z + e1.z + e2.z; m2 = m2 > 0.f ? m2 : 0.f;
    float m3 = hv.w + e0.w + e1.w + e2.w; m3 = m3 > 0.f ? m3 : 0.f;
    float* o = agg + (size_t)dst * HID + c;
    atomicAdd(o + 0, m0);
    atomicAdd(o + 1, m1);
    atomicAdd(o + 2, m2);
    atomicAdd(o + 3, m3);
}

// ---------------------------------------------------------------------------
// Pre-convert lin1_w/lin2_w (f32 [L,128,128], K-major x N) into fragment-
// ordered bf16 so each lane's 16-element B fragment is one contiguous 32B.
// ISA 16-bit B layout (32x16, wave32): lane holds column N=lane&15;
// element j<8 -> K = 8*(lane>>4) + j ; j>=8 -> K = 16 + 8*(lane>>4) + (j-8)
// Frag index: [layer][kstep 0..3][ntile 0..7][lane 0..31][j 0..15]
// ---------------------------------------------------------------------------
__global__ __launch_bounds__(256) void k_prep_weights(
    const float* __restrict__ w1, const float* __restrict__ w2,
    __bf16* __restrict__ f1, __bf16* __restrict__ f2)
{
    int o = blockIdx.x * 256 + threadIdx.x;      // 4*4*8*32*16 = 65536 total
    if (o >= NLAYERS * 4 * 8 * 32 * 16) return;
    int j     =  o        & 15;
    int lane  = (o >> 4)  & 31;
    int ntile = (o >> 9)  & 7;
    int ks    = (o >> 12) & 3;
    int layer =  o >> 14;
    int colL  = lane & 15;
    int khalf = (lane >> 4) * 8;
    int K     = ks * 32 + ((j < 8) ? (khalf + j) : (16 + khalf + (j - 8)));
    int col   = ntile * 16 + colL;
    size_t src = (size_t)layer * HID * HID + (size_t)K * HID + col;
    size_t dst = (size_t)layer * 16384 +
                 (((size_t)ks * 8 + ntile) * 32 + lane) * 16 + j;
    f1[dst] = (__bf16)w1[src];
    f2[dst] = (__bf16)w2[src];
}

// ---------------------------------------------------------------------------
// Fused node update: h_next = h + relu(BN(relu(z@W1+b1)@W2+b2))
// Block = 256 threads = 8 waves; block tile = 16 nodes x 128 channels.
// Wave w computes the 16x16 output tile for columns [16w, 16w+16).
// Both GEMMs via V_WMMA_F32_16X16X32_BF16; K loop = 4; Z1 staged in LDS.
// ---------------------------------------------------------------------------
__global__ __launch_bounds__(256) void k_node_update(
    const float* __restrict__ agg, const float* __restrict__ hcur,
    const __bf16* __restrict__ wf1, const __bf16* __restrict__ wf2,
    const float* __restrict__ b1, const float* __restrict__ b2,
    const float* __restrict__ gamma, const float* __restrict__ beta,
    const float* __restrict__ mean, const float* __restrict__ var,
    float* __restrict__ hnext)
{
    __shared__ __align__(16) __bf16 zt[16 * HID];
    __shared__ __align__(16) __bf16 z1[16 * HID];

    const int tid   = threadIdx.x;
    const int wave  = tid >> 5;
    const int lane  = tid & 31;
    const int node0 = blockIdx.x * 16;

    // stage z = h + agg tile into LDS as bf16 (8 values per thread)
    {
        const int base = tid * 8;
        const int row  = base >> 7;
        const int c    = base & 127;
        const float4* p = (const float4*)(agg + (size_t)(node0 + row) * HID + c);
        float4 v0 = p[0], v1 = p[1];
        __bf16* q = &zt[base];
        q[0] = (__bf16)v0.x; q[1] = (__bf16)v0.y;
        q[2] = (__bf16)v0.z; q[3] = (__bf16)v0.w;
        q[4] = (__bf16)v1.x; q[5] = (__bf16)v1.y;
        q[6] = (__bf16)v1.z; q[7] = (__bf16)v1.w;
    }
    __syncthreads();

    const int colL  = lane & 15;
    const int col   = wave * 16 + colL;
    const int khalf = (lane >> 4) * 8;   // A/B half-wave K-octet select
    const int rowA  = lane & 15;         // A-matrix: both half-waves hold M=0..15

    // ---- GEMM 1: acc1 = z @ W1 ----
    v8f acc1 = {};
    #pragma unroll
    for (int ks = 0; ks < 4; ++ks) {
        const __bf16* ap = &zt[rowA * HID + ks * 32 + khalf];
        v8bf alo = *(const v8bf*)(ap);
        v8bf ahi = *(const v8bf*)(ap + 16);
        v16bf a = __builtin_shufflevector(alo, ahi,
                    0,1,2,3,4,5,6,7,8,9,10,11,12,13,14,15);
        v16bf b = *(const v16bf*)(wf1 + (((size_t)ks * 8 + wave) * 32 + lane) * 16);
        acc1 = __builtin_amdgcn_wmma_f32_16x16x32_bf16(
                   false, a, false, b, (short)0, acc1, false, false);
    }

    // bias + relu -> Z1 tile in LDS (bf16), C/D layout: lane=(col), vgpr r = row
    const float bias1 = b1[col];
    #pragma unroll
    for (int r = 0; r < 8; ++r) {
        int m = (lane < 16) ? r : (r + 8);
        float v = acc1[r] + bias1;
        v = v > 0.f ? v : 0.f;
        z1[m * HID + col] = (__bf16)v;
    }
    __syncthreads();

    // ---- GEMM 2: acc2 = Z1 @ W2 ----
    v8f acc2 = {};
    #pragma unroll
    for (int ks = 0; ks < 4; ++ks) {
        const __bf16* ap = &z1[rowA * HID + ks * 32 + khalf];
        v8bf alo = *(const v8bf*)(ap);
        v8bf ahi = *(const v8bf*)(ap + 16);
        v16bf a = __builtin_shufflevector(alo, ahi,
                    0,1,2,3,4,5,6,7,8,9,10,11,12,13,14,15);
        v16bf b = *(const v16bf*)(wf2 + (((size_t)ks * 8 + wave) * 32 + lane) * 16);
        acc2 = __builtin_amdgcn_wmma_f32_16x16x32_bf16(
                   false, a, false, b, (short)0, acc2, false, false);
    }

    // bias2 + BN + relu + residual, store f32
    const float bias2 = b2[col];
    const float mu    = mean[col];
    const float inv   = rsqrtf(var[col] + 1e-5f);
    const float ga    = gamma[col];
    const float be    = beta[col];
    #pragma unroll
    for (int r = 0; r < 8; ++r) {
        int m = (lane < 16) ? r : (r + 8);
        size_t idx = (size_t)(node0 + m) * HID + col;
        float v = acc2[r] + bias2;
        v = (v - mu) * inv * ga + be;
        v = v > 0.f ? v : 0.f;
        hnext[idx] = hcur[idx] + v;
    }
}

// ---------------------------------------------------------------------------
// Pooling
// ---------------------------------------------------------------------------
__global__ __launch_bounds__(256) void k_pool_zero(
    float* __restrict__ sums, float* __restrict__ cnts)
{
    int i = blockIdx.x * 256 + threadIdx.x;
    if (i < N_GRAPHS * HID) sums[i] = 0.f;
    if (i < N_GRAPHS)       cnts[i] = 0.f;
}

__global__ __launch_bounds__(256) void k_pool_acc(
    const float* __restrict__ h, const int* __restrict__ batch,
    float* __restrict__ sums, float* __restrict__ cnts)
{
    int gid = blockIdx.x * 256 + threadIdx.x;
    if (gid >= N_NODES * 32) return;
    int n = gid >> 5;
    int c = (gid & 31) * 4;
    int g = batch[n];
    const float4 v = *(const float4*)(h + (size_t)n * HID + c);
    float* o = sums + (size_t)g * HID + c;
    atomicAdd(o + 0, v.x);
    atomicAdd(o + 1, v.y);
    atomicAdd(o + 2, v.z);
    atomicAdd(o + 3, v.w);
    if ((gid & 31) == 0) atomicAdd(cnts + g, 1.0f);
}

__global__ __launch_bounds__(128) void k_pool_mlp(
    const float* __restrict__ sums, const float* __restrict__ cnts,
    const float* __restrict__ mw, const float* __restrict__ mb,
    float* __restrict__ out)
{
    __shared__ float p[HID];
    int g = blockIdx.x;
    int t = threadIdx.x;
    float cnt = cnts[g];
    cnt = cnt > 1.f ? cnt : 1.f;
    p[t] = sums[(size_t)g * HID + t] / cnt;
    __syncthreads();
    if (t < OUTD) {
        float a = mb[t];
        #pragma unroll 16
        for (int c = 0; c < HID; ++c) a += p[c] * mw[c * OUTD + t];
        out[g * OUTD + t] = a;
    }
}

// ---------------------------------------------------------------------------
extern "C" void kernel_launch(void* const* d_in, const int* in_sizes, int n_in,
                              void* d_out, int out_size, void* d_ws, size_t ws_size,
                              hipStream_t stream) {
    const int*   x          = (const int*)  d_in[0];
    const int*   edge_index = (const int*)  d_in[1];
    const int*   edge_attr  = (const int*)  d_in[2];
    const int*   batch      = (const int*)  d_in[3];
    const float* atom_emb   = (const float*)d_in[4];
    const float* bond_emb   = (const float*)d_in[5];
    const float* lin1_w     = (const float*)d_in[6];
    const float* lin1_b     = (const float*)d_in[7];
    const float* lin2_w     = (const float*)d_in[8];
    const float* lin2_b     = (const float*)d_in[9];
    const float* bn_gamma   = (const float*)d_in[10];
    const float* bn_beta    = (const float*)d_in[11];
    const float* bn_mean    = (const float*)d_in[12];
    const float* bn_var     = (const float*)d_in[13];
    const float* mlp_w      = (const float*)d_in[14];
    const float* mlp_b      = (const float*)d_in[15];

    char* ws = (char*)d_ws;
    size_t off = 0;
    auto alloc = [&](size_t bytes) -> char* {
        char* p = ws + off;
        off = (off + bytes + 255) & ~(size_t)255;
        return p;
    };
    float*  h0   = (float*) alloc((size_t)N_NODES * HID * 4);
    float*  h1   = (float*) alloc((size_t)N_NODES * HID * 4);
    float*  agg  = (float*) alloc((size_t)N_NODES * HID * 4);
    __bf16* wf1  = (__bf16*)alloc((size_t)NLAYERS * 16384 * 2);
    __bf16* wf2  = (__bf16*)alloc((size_t)NLAYERS * 16384 * 2);
    float*  sums = (float*) alloc((size_t)N_GRAPHS * HID * 4);
    float*  cnts = (float*) alloc((size_t)N_GRAPHS * 4);

    const int nodeChunks = N_NODES * 32;   // (node, 4-ch) work items
    const int edgeChunks = N_EDGES * 32;

    k_prep_weights<<<(NLAYERS * 4 * 8 * 32 * 16 + 255) / 256, 256, 0, stream>>>(
        lin1_w, lin2_w, wf1, wf2);
    k_encode_nodes<<<(nodeChunks + 255) / 256, 256, 0, stream>>>(x, atom_emb, h0);

    float* hc = h0;
    float* hn = h1;
    for (int i = 0; i < NLAYERS; ++i) {
        k_copy_h<<<(nodeChunks + 255) / 256, 256, 0, stream>>>(hc, agg);
        k_edges<<<(edgeChunks + 255) / 256, 256, 0, stream>>>(
            edge_index, edge_attr, bond_emb, hc, agg);
        k_node_update<<<N_NODES / 16, 256, 0, stream>>>(
            agg, hc, wf1 + (size_t)i * 16384, wf2 + (size_t)i * 16384,
            lin1_b + i * HID, lin2_b + i * HID,
            bn_gamma + i * HID, bn_beta + i * HID,
            bn_mean + i * HID, bn_var + i * HID, hn);
        float* t = hc; hc = hn; hn = t;
    }

    k_pool_zero<<<(N_GRAPHS * HID + 255) / 256, 256, 0, stream>>>(sums, cnts);
    k_pool_acc<<<(nodeChunks + 255) / 256, 256, 0, stream>>>(hc, batch, sums, cnts);
    k_pool_mlp<<<N_GRAPHS, 128, 0, stream>>>(sums, cnts, mlp_w, mlp_b, (float*)d_out);
}
